// Estimator_1606317769382
// MI455X (gfx1250) — matrix operand order repaired and checked by reading, not compiled, not run on hardware
//
#include <hip/hip_runtime.h>
#include <math.h>
#include <stdint.h>

typedef __bf16 bf16;
typedef __attribute__((ext_vector_type(16))) __bf16 v16bf;
typedef __attribute__((ext_vector_type(8)))  __bf16 v8bf;
typedef __attribute__((ext_vector_type(8)))  float  v8f;

#define T_STEPS 8192
#define D_IN    1024
#define H_DIM   1024
#define D_OUT   1024
#define N_OUT   2048

// ---------------------------------------------------------------------------
// fp32 -> bf16 conversion (grid-stride)
// ---------------------------------------------------------------------------
__global__ void __launch_bounds__(256) cvt_f32_bf16(const float* __restrict__ in,
                                                    bf16* __restrict__ out, int n) {
    for (int i = blockIdx.x * blockDim.x + threadIdx.x; i < n; i += gridDim.x * blockDim.x)
        out[i] = (bf16)in[i];
}

// ---------------------------------------------------------------------------
// Fragment loaders following the CDNA5 ISA 16-bit A (16x32) / B (32x16) tables.
//   A: lane half h holds K = {k0+8h..+7, k0+16+8h..+7}
//   B: lane half h holds K = {k0+8h..+7, k0+16+8h..+7}
// ---------------------------------------------------------------------------
__device__ __forceinline__ v16bf load_afrag(const bf16* row, int k0, int half) {
    v8bf lo = *(const v8bf*)(row + k0 + half * 8);
    v8bf hi = *(const v8bf*)(row + k0 + 16 + half * 8);
    return __builtin_shufflevector(lo, hi, 0,1,2,3,4,5,6,7, 8,9,10,11,12,13,14,15);
}
__device__ __forceinline__ v16bf load_bfrag(const bf16* row, int k0, int half) {
    v8bf lo = *(const v8bf*)(row + k0 + half * 8);
    v8bf hi = *(const v8bf*)(row + k0 + 16 + half * 8);
    return __builtin_shufflevector(lo, hi, 0,1,2,3,4,5,6,7, 8,9,10,11,12,13,14,15);
}

// ---------------------------------------------------------------------------
// C[M,N] = A[M,K] @ B[N,K]^T (+ bias[N]); A,B bf16 row-major, C fp32.
// 256 threads = 8 waves as 2(m) x 4(n); each wave computes a 32x32 tile
// (2x2 WMMA tiles, A and B fragments each reused twice -> 4 WMMAs / 8 B128
// loads per k-tile). Fragments double-buffered in registers so the next
// tile's loads overlap the current tile's WMMAs.
// ---------------------------------------------------------------------------
__global__ void __launch_bounds__(256)
gemm_bf16_nt(const bf16* __restrict__ A, const bf16* __restrict__ B,
             const float* __restrict__ bias, float* __restrict__ C,
             int M, int N, int K)
{
    const int lane = threadIdx.x & 31;
    const int wave = threadIdx.x >> 5;        // 0..7
    const int wm   = wave >> 2;               // 0..1
    const int wn   = wave & 3;                // 0..3
    const int m0   = blockIdx.x * 64  + wm * 32;   // two m tiles: m0, m0+16
    const int n0   = blockIdx.y * 128 + wn * 32;   // two n tiles: n0, n0+16
    const int half = lane >> 4;               // 0 | 1
    const int l16  = lane & 15;

    const bf16* arow0 = A + (size_t)(m0 + l16) * K;
    const bf16* arow1 = A + (size_t)(m0 + 16 + l16) * K;
    const bf16* brow0 = B + (size_t)(n0 + l16) * K;
    const bf16* brow1 = B + (size_t)(n0 + 16 + l16) * K;

    v8f c00 = {}, c01 = {}, c10 = {}, c11 = {};

    v16bf a0 = load_afrag(arow0, 0, half);
    v16bf a1 = load_afrag(arow1, 0, half);
    v16bf b0 = load_bfrag(brow0, 0, half);
    v16bf b1 = load_bfrag(brow1, 0, half);

    for (int k0 = 0; k0 < K - 32; k0 += 32) {
        const int kn = k0 + 32;
        v16bf na0 = load_afrag(arow0, kn, half);
        v16bf na1 = load_afrag(arow1, kn, half);
        v16bf nb0 = load_bfrag(brow0, kn, half);
        v16bf nb1 = load_bfrag(brow1, kn, half);

        c00 = __builtin_amdgcn_wmma_f32_16x16x32_bf16(false, a0, false, b0, (short)0, c00, false, false);
        c01 = __builtin_amdgcn_wmma_f32_16x16x32_bf16(false, a0, false, b1, (short)0, c01, false, false);
        c10 = __builtin_amdgcn_wmma_f32_16x16x32_bf16(false, a1, false, b0, (short)0, c10, false, false);
        c11 = __builtin_amdgcn_wmma_f32_16x16x32_bf16(false, a1, false, b1, (short)0, c11, false, false);

        a0 = na0; a1 = na1; b0 = nb0; b1 = nb1;
    }
    c00 = __builtin_amdgcn_wmma_f32_16x16x32_bf16(false, a0, false, b0, (short)0, c00, false, false);
    c01 = __builtin_amdgcn_wmma_f32_16x16x32_bf16(false, a0, false, b1, (short)0, c01, false, false);
    c10 = __builtin_amdgcn_wmma_f32_16x16x32_bf16(false, a1, false, b0, (short)0, c10, false, false);
    c11 = __builtin_amdgcn_wmma_f32_16x16x32_bf16(false, a1, false, b1, (short)0, c11, false, false);

    // C layout: VGPR r -> row (tile_m + r + 8*half); col = (tile_n + l16)
    const float bv0 = bias ? bias[n0 + l16]      : 0.f;
    const float bv1 = bias ? bias[n0 + 16 + l16] : 0.f;
    const int   mrA = m0 + half * 8;
    const int   mrB = m0 + 16 + half * 8;
    #pragma unroll
    for (int r = 0; r < 8; ++r) {
        const size_t rA = (size_t)(mrA + r) * N;
        const size_t rB = (size_t)(mrB + r) * N;
        C[rA + n0 + l16]      = c00[r] + bv0;
        C[rA + n0 + 16 + l16] = c01[r] + bv1;
        C[rB + n0 + l16]      = c10[r] + bv0;
        C[rB + n0 + 16 + l16] = c11[r] + bv1;
    }
}

// ---------------------------------------------------------------------------
// Persistent RNN scan. NB blocks, each owns ROWS rows of W_hh in LDS (fp32).
// Per step: y[r] = dot(W_hh[r,:], h) by 8 threads/row from LDS, then
// tanh(xp + y) -> hs[t+1]; grid barrier via monotone atomic counter.
// LDS: 32*1024*4 (weights) + 1024*4 (h) = 132 KB of the 320 KB WGP pool.
// ---------------------------------------------------------------------------
#define NB   32
#define ROWS 32

__global__ void __launch_bounds__(256)
rnn_scan(const float* __restrict__ W_hh, const float* __restrict__ xp,
         float* __restrict__ hs, unsigned* __restrict__ syncc)
{
    extern __shared__ float smem[];
    float* Wl  = smem;                 // [ROWS][H_DIM]
    float* hsh = smem + ROWS * H_DIM;  // [H_DIM]

    const int tid = threadIdx.x;
    const int r0  = blockIdx.x * ROWS;

    for (int i = tid; i < ROWS * H_DIM; i += 256)
        Wl[i] = W_hh[(size_t)r0 * H_DIM + i];

    const int row   = tid >> 3;        // 0..31
    const int sub   = tid & 7;         // 0..7
    const int kbase = sub * (H_DIM / 8);

    for (int t = 0; t < T_STEPS; ++t) {
        __syncthreads();
        for (int i = tid; i < H_DIM; i += 256)
            hsh[i] = hs[(size_t)t * H_DIM + i];
        __syncthreads();

        const float* wr = Wl + row * H_DIM + kbase;
        const float* hv = hsh + kbase;
        float acc = 0.f;
        #pragma unroll 8
        for (int k = 0; k < H_DIM / 8; ++k)
            acc = fmaf(wr[k], hv[k], acc);

        acc += __shfl_xor(acc, 1, 32);
        acc += __shfl_xor(acc, 2, 32);
        acc += __shfl_xor(acc, 4, 32);

        if (sub == 0) {
            float v = tanhf(xp[(size_t)t * H_DIM + r0 + row] + acc);
            hs[(size_t)(t + 1) * H_DIM + r0 + row] = v;
        }

        __threadfence();
        __syncthreads();
        if (tid == 0) {
            __hip_atomic_fetch_add(syncc, 1u, __ATOMIC_ACQ_REL, __HIP_MEMORY_SCOPE_AGENT);
            const unsigned target = (unsigned)(t + 1) * NB;
            while (__hip_atomic_load(syncc, __ATOMIC_ACQUIRE, __HIP_MEMORY_SCOPE_AGENT) < target)
                __builtin_amdgcn_s_sleep(1);
        }
        __syncthreads();
    }
}

// ---------------------------------------------------------------------------
// init: zero the sync counter + h0 row (deterministic across graph replays)
// ---------------------------------------------------------------------------
__global__ void __launch_bounds__(256) init_state(float* hs_row0, unsigned* syncc) {
    int i = blockIdx.x * blockDim.x + threadIdx.x;
    if (i == 0) *syncc = 0u;
    if (i < H_DIM) hs_row0[i] = 0.f;
}

// ---------------------------------------------------------------------------
// gather selected hidden rows, convert to bf16 A-matrix
// ---------------------------------------------------------------------------
__global__ void __launch_bounds__(256)
gather_bf16(const float* __restrict__ hs, const int* __restrict__ idx,
            bf16* __restrict__ out) {
    const int m = blockIdx.x;
    const int t = idx[m];
    const float* src = hs + (size_t)(t + 1) * H_DIM;
    bf16* dst = out + (size_t)m * H_DIM;
    for (int i = threadIdx.x; i < H_DIM; i += 256) dst[i] = (bf16)src[i];
}

// ---------------------------------------------------------------------------
// row softmax (fp32 logits -> bf16 probs), one block per row
// ---------------------------------------------------------------------------
__global__ void __launch_bounds__(256)
softmax_bf16(const float* __restrict__ in, bf16* __restrict__ out, int N) {
    __shared__ float redm[8];
    __shared__ float reds[8];
    const int row  = blockIdx.x;
    const int lane = threadIdx.x & 31;
    const int wave = threadIdx.x >> 5;
    const float* x = in + (size_t)row * N;

    float mx = -INFINITY;
    for (int i = threadIdx.x; i < N; i += 256) mx = fmaxf(mx, x[i]);
    for (int o = 16; o > 0; o >>= 1) mx = fmaxf(mx, __shfl_xor(mx, o, 32));
    if (lane == 0) redm[wave] = mx;
    __syncthreads();
    mx = redm[0];
    #pragma unroll
    for (int w = 1; w < 8; ++w) mx = fmaxf(mx, redm[w]);

    float s = 0.f;
    for (int i = threadIdx.x; i < N; i += 256) s += expf(x[i] - mx);
    for (int o = 16; o > 0; o >>= 1) s += __shfl_xor(s, o, 32);
    if (lane == 0) reds[wave] = s;
    __syncthreads();
    s = 0.f;
    #pragma unroll
    for (int w = 0; w < 8; ++w) s += reds[w];
    const float inv = 1.f / s;

    for (int i = threadIdx.x; i < N; i += 256)
        out[(size_t)row * N + i] = (bf16)(expf(x[i] - mx) * inv);
}

// ---------------------------------------------------------------------------
extern "C" void kernel_launch(void* const* d_in, const int* in_sizes, int n_in,
                              void* d_out, int out_size, void* d_ws, size_t ws_size,
                              hipStream_t stream) {
    const float* x     = (const float*)d_in[0];
    const float* W_ih  = (const float*)d_in[1];
    const float* W_hh  = (const float*)d_in[2];
    const float* b_h   = (const float*)d_in[3];
    const float* W_ho  = (const float*)d_in[4];
    const float* b_o   = (const float*)d_in[5];
    const float* W_fc  = (const float*)d_in[6];
    const float* b_fc  = (const float*)d_in[7];
    const int*   oidx  = (const int*)d_in[8];
    float*       out   = (float*)d_out;

    char* ws = (char*)d_ws;
    size_t off = 0;
    auto carve = [&](size_t bytes) -> void* {
        void* p = ws + off;
        off = (off + bytes + 255) & ~(size_t)255;
        return p;
    };

    bf16*     x_bf    = (bf16*)    carve((size_t)T_STEPS * D_IN  * 2);
    bf16*     Wih_bf  = (bf16*)    carve((size_t)H_DIM   * D_IN  * 2);
    bf16*     Who_bf  = (bf16*)    carve((size_t)D_OUT   * H_DIM * 2);
    bf16*     Wfc_bf  = (bf16*)    carve((size_t)D_OUT   * D_OUT * 2);
    float*    xp      = (float*)   carve((size_t)T_STEPS * H_DIM * 4);
    float*    hs      = (float*)   carve((size_t)(T_STEPS + 1) * H_DIM * 4);
    bf16*     Asel    = (bf16*)    carve((size_t)N_OUT   * H_DIM * 2);
    float*    logits  = (float*)   carve((size_t)N_OUT   * D_OUT * 4);
    bf16*     probs   = (bf16*)    carve((size_t)N_OUT   * D_OUT * 2);
    unsigned* syncc   = (unsigned*)carve(256);
    (void)ws_size; (void)in_sizes; (void)n_in; (void)out_size;

    // 0) deterministic state reset (counter + h0)
    init_state<<<dim3((H_DIM + 255) / 256), dim3(256), 0, stream>>>(hs, syncc);

    // 1) fp32 -> bf16 for WMMA operands
    cvt_f32_bf16<<<dim3(2048), dim3(256), 0, stream>>>(x,    x_bf,   T_STEPS * D_IN);
    cvt_f32_bf16<<<dim3(1024), dim3(256), 0, stream>>>(W_ih, Wih_bf, H_DIM * D_IN);
    cvt_f32_bf16<<<dim3(1024), dim3(256), 0, stream>>>(W_ho, Who_bf, D_OUT * H_DIM);
    cvt_f32_bf16<<<dim3(1024), dim3(256), 0, stream>>>(W_fc, Wfc_bf, D_OUT * D_OUT);

    // 2) xp = x @ W_ih^T + b_h   [8192,1024] (WMMA bf16)
    gemm_bf16_nt<<<dim3(T_STEPS / 64, H_DIM / 128), dim3(256), 0, stream>>>(
        x_bf, Wih_bf, b_h, xp, T_STEPS, H_DIM, D_IN);

    // 3) sequential scan, W_hh resident in LDS, grid barrier per step
    const size_t scan_lds = (size_t)(ROWS * H_DIM + H_DIM) * sizeof(float); // 132 KB
    rnn_scan<<<dim3(NB), dim3(256), scan_lds, stream>>>(W_hh, xp, hs, syncc);

    // 4) gather selected steps -> bf16
    gather_bf16<<<dim3(N_OUT), dim3(256), 0, stream>>>(hs, oidx, Asel);

    // 5) logits = h_sel @ W_ho^T + b_o   (WMMA bf16)
    gemm_bf16_nt<<<dim3(N_OUT / 64, D_OUT / 128), dim3(256), 0, stream>>>(
        Asel, Who_bf, b_o, logits, N_OUT, D_OUT, H_DIM);

    // 6) softmax rows -> bf16 probs
    softmax_bf16<<<dim3(N_OUT), dim3(256), 0, stream>>>(logits, probs, D_OUT);

    // 7) out = probs @ W_fc^T + b_fc   (WMMA bf16)
    gemm_bf16_nt<<<dim3(N_OUT / 64, D_OUT / 128), dim3(256), 0, stream>>>(
        probs, Wfc_bf, b_fc, out, N_OUT, D_OUT, D_OUT);
}
